// Denoise_14929306321386
// MI455X (gfx1250) — matrix operand order repaired and checked by reading, not compiled, not run on hardware
//
#include <hip/hip_runtime.h>
#include <stdint.h>

typedef unsigned short u16t;
typedef __attribute__((ext_vector_type(16))) __bf16          v16bf;
typedef __attribute__((ext_vector_type(16))) unsigned short  v16u;
typedef __attribute__((ext_vector_type(8)))  float           v8f;

#define N_    50000
#define F_    64
#define E_TOT 1000000
#define L_    3
#define WV    4   // waves (16-row tiles) per 128-thread block

// ---------- small helpers ----------
__device__ __forceinline__ u16t f2bf(float f) {
  unsigned u = __float_as_uint(f);
  unsigned r = u + 0x7fffu + ((u >> 16) & 1u);   // round-to-nearest-even
  return (u16t)(r >> 16);
}
__device__ __forceinline__ float bf2f(u16t h) {
  return __uint_as_float(((unsigned)h) << 16);
}
__device__ __forceinline__ float silu(float v) { return v / (1.f + __expf(-v)); }

__device__ __forceinline__ v8f wmma_bf(v16u a, v16u b, v8f c) {
  return __builtin_amdgcn_wmma_f32_16x16x32_bf16(
      false, __builtin_bit_cast(v16bf, a),
      false, __builtin_bit_cast(v16bf, b),
      (short)0, c, false, false);
}
__device__ __forceinline__ void zacc(v8f c[4]) {
#pragma unroll
  for (int i = 0; i < 4; ++i)
#pragma unroll
    for (int j = 0; j < 8; ++j) c[i][j] = 0.f;
}

// A tile: LDS [16][W] bf16 row-major.  Weights: pre-swizzled B fragments,
// 512 u16 per (kstep,ntile) block, laid out [lane][j] so each lane does one
// contiguous 32B load.  ISA A-layout: lane holds row m=lane&15,
// element j -> k = (j&7) + 16*(j>>3) + 8*(lane>>4).
__device__ __forceinline__ void wgemm(const u16t* ta, int W, int ksteps,
                                      const u16t* wf, v8f c[4], int lane) {
  const int row   = lane & 15;
  const int khalf = (lane >> 4) << 3;
  for (int ks = 0; ks < ksteps; ++ks) {
    v16u a;
    const int kb = ks * 32 + khalf;
#pragma unroll
    for (int j = 0; j < 16; ++j) {
      int k = kb + (j & 7) + ((j >> 3) << 4);
      a[j]  = ta[row * W + k];
    }
#pragma unroll
    for (int nt = 0; nt < 4; ++nt) {
      const v16u* bp = (const v16u*)(wf + ((size_t)(ks * 4 + nt) << 9));
      v16u b = bp[lane];
      c[nt]  = wmma_bf(a, b, c[nt]);
    }
  }
}

// C/D layout: element r of lane -> m = r + 8*(lane>>4), n = (lane&15) + 16*nt
__device__ __forceinline__ void epi_silu(v8f c[4], const float* bias,
                                         u16t* dst, int lane) {
  const int n0 = lane & 15, mh = (lane >> 4) << 3;
#pragma unroll
  for (int nt = 0; nt < 4; ++nt) {
    float bn = bias[nt * 16 + n0];
#pragma unroll
    for (int r = 0; r < 8; ++r) {
      float v = c[nt][r] + bn;
      dst[(mh + r) * 64 + nt * 16 + n0] = f2bf(silu(v));
    }
  }
}

// ---------- utility kernels ----------
__global__ void k_cvt(const float* __restrict__ src, u16t* __restrict__ dst, int K) {
  int idx   = blockIdx.x * blockDim.x + threadIdx.x;
  int total = K * 64;
  if (idx >= total) return;
  int frag = idx >> 9, rem = idx & 511;
  int lane = rem >> 4, j = rem & 15;
  int ks = frag >> 2, nt = frag & 3;
  int k = ks * 32 + ((lane >> 4) << 3) + (j & 7) + ((j >> 3) << 4);   // B layout: k = (j&7) + 16*(j>>3) + 8*(lane>>4)
  int n = nt * 16 + (lane & 15);
  dst[idx] = f2bf(src[(size_t)k * 64 + n]);
}

__global__ void k_embed(const float* __restrict__ emb, const int* __restrict__ z,
                        const float* __restrict__ x, float* __restrict__ h,
                        float* __restrict__ xcur) {
  int idx = blockIdx.x * blockDim.x + threadIdx.x;
  if (idx < N_ * F_) {
    int i = idx >> 6, f = idx & 63;
    h[idx] = emb[(size_t)z[i] * F_ + f];
  }
  if (idx < N_ * 3) xcur[idx] = x[idx];
}

__global__ void k_zero32(unsigned* __restrict__ p, long n) {
  long idx = (long)blockIdx.x * blockDim.x + threadIdx.x;
  if (idx < n) p[idx] = 0u;
}
__global__ void k_zero_f(float* __restrict__ p, int n) {
  int idx = blockIdx.x * blockDim.x + threadIdx.x;
  if (idx < n) p[idx] = 0.f;
}
__global__ void k_copy(float* __restrict__ dst, const float* __restrict__ src, int n) {
  int idx = blockIdx.x * blockDim.x + threadIdx.x;
  if (idx < n) dst[idx] = src[idx];
}

// ---------- edge kernel: one wave32 per 16-edge tile ----------
__global__ __launch_bounds__(128) void k_edge(
    const int* __restrict__ edges, const float* __restrict__ xcur,
    const float* __restrict__ h, u16t* __restrict__ hij,
    const u16t* __restrict__ fr_eu1, const float* __restrict__ eu_b1,
    const u16t* __restrict__ fr_eu2, const float* __restrict__ eu_b2,
    const u16t* __restrict__ fr_m1,  const float* __restrict__ m_b1,
    const u16t* __restrict__ fr_m2,  const float* __restrict__ m_b2,
    const u16t* __restrict__ fr_p1,  const float* __restrict__ p_b1,
    const float* __restrict__ m_w1_last,
    const float* __restrict__ p_w2,  const float* __restrict__ p_b2,
    float* __restrict__ sum_m, float* __restrict__ sum_xw,
    float* __restrict__ cnt, int Etot) {
  __shared__ u16t  s_in[WV][16 * 192];
  __shared__ u16t  s_a [WV][16 * 64];
  __shared__ u16t  s_b [WV][16 * 64];
  __shared__ float s_d [WV][16];
  __shared__ int   s_e0[WV][16];
  __shared__ int   s_e1[WV][16];

  const int lane  = threadIdx.x & 31;
  const int w     = threadIdx.x >> 5;
  const int tile  = blockIdx.x * WV + w;
  const int ebase = tile * 16;
  const bool act  = ebase < Etot;
  const int n0 = lane & 15, mh = (lane >> 4) << 3;

  if (act && lane < 16) {
    int e = ebase + lane;
    int a = edges[e], b = edges[Etot + e];
    s_e0[w][lane] = a;
    s_e1[w][lane] = b;
    float dx = xcur[3 * b + 0] - xcur[3 * a + 0];
    float dy = xcur[3 * b + 1] - xcur[3 * a + 1];
    float dz = xcur[3 * b + 2] - xcur[3 * a + 2];
    s_d[w][lane] = sqrtf(dx * dx + dy * dy + dz * dz);
  }
  __syncthreads();

  // stage concat(h[e0], h[e1], h_ij) as bf16 [16][192]
  if (act) {
    for (int m = 0; m < 16; ++m) {
      int ea = s_e0[w][m], eb = s_e1[w][m];
      size_t eh = (size_t)(ebase + m) * 64;
      for (int cc = lane; cc < 192; cc += 32) {
        u16t v;
        if (cc < 64)       v = f2bf(h[(size_t)ea * 64 + cc]);
        else if (cc < 128) v = f2bf(h[(size_t)eb * 64 + (cc - 64)]);
        else               v = hij[eh + (cc - 128)];
        s_in[w][m * 192 + cc] = v;
      }
    }
  }
  __syncthreads();

  v8f c[4];

  // EdgeUpdate layer 1: [16,192] x [192,64] -> SiLU
  if (act) {
    zacc(c);
    wgemm(&s_in[w][0], 192, 6, fr_eu1, c, lane);
    epi_silu(c, eu_b1, &s_a[w][0], lane);
  }
  __syncthreads();

  // EdgeUpdate layer 2 + residual into h_ij (LDS + global, bf16)
  if (act) {
    zacc(c);
    wgemm(&s_a[w][0], 64, 2, fr_eu2, c, lane);
#pragma unroll
    for (int nt = 0; nt < 4; ++nt) {
      float bn = eu_b2[nt * 16 + n0];
#pragma unroll
      for (int r = 0; r < 8; ++r) {
        int m = mh + r, n = nt * 16 + n0;
        float v = bf2f(s_in[w][m * 192 + 128 + n]) + c[nt][r] + bn;
        u16t hv = f2bf(v);
        s_in[w][m * 192 + 128 + n] = hv;
        hij[(size_t)(ebase + m) * 64 + n] = hv;
      }
    }
  }
  __syncthreads();

  // Message layer 1: K=192 WMMA + d_ij rank-1 (193rd input column) + SiLU
  if (act) {
    zacc(c);
    wgemm(&s_in[w][0], 192, 6, fr_m1, c, lane);
#pragma unroll
    for (int nt = 0; nt < 4; ++nt) {
      int n = nt * 16 + n0;
      float bn = m_b1[n], wl = m_w1_last[n];
#pragma unroll
      for (int r = 0; r < 8; ++r) {
        int m = mh + r;
        float v = c[nt][r] + bn + s_d[w][m] * wl;
        s_a[w][m * 64 + n] = f2bf(silu(v));
      }
    }
  }
  __syncthreads();

  // Message layer 2 -> m_ij; scatter into sum_m via L2 atomics
  if (act) {
    zacc(c);
    wgemm(&s_a[w][0], 64, 2, fr_m2, c, lane);
#pragma unroll
    for (int nt = 0; nt < 4; ++nt) {
      int n = nt * 16 + n0;
      float bn = m_b2[n];
#pragma unroll
      for (int r = 0; r < 8; ++r) {
        int m = mh + r;
        float mi = silu(c[nt][r] + bn);
        s_b[w][m * 64 + n] = f2bf(mi);
        atomicAdd(&sum_m[(size_t)s_e0[w][m] * 64 + n], mi);
      }
    }
  }
  __syncthreads();

  // PosUpdate layer 1: [16,64] x [64,64] -> SiLU
  if (act) {
    zacc(c);
    wgemm(&s_b[w][0], 64, 2, fr_p1, c, lane);
    epi_silu(c, p_b1, &s_a[w][0], lane);
  }
  __syncthreads();

  // PosUpdate layer 2 (64->1, VALU dot) + position segment sums
  if (act && lane < 16) {
    int m = lane;
    float acc = p_b2[0];
#pragma unroll 8
    for (int k = 0; k < 64; ++k) acc += bf2f(s_a[w][m * 64 + k]) * p_w2[k];
    int si = s_e0[w][m], ti = s_e1[w][m];
#pragma unroll
    for (int d3 = 0; d3 < 3; ++d3) {
      float dv = xcur[3 * ti + d3] - xcur[3 * si + d3];
      atomicAdd(&sum_xw[3 * si + d3], dv * acc);
    }
    atomicAdd(&cnt[si], 1.0f);
  }
}

// ---------- node kernel: segment means + node MLP (WMMA) ----------
__global__ __launch_bounds__(128) void k_node(
    float* __restrict__ h, float* __restrict__ xcur,
    const float* __restrict__ sum_m, const float* __restrict__ sum_xw,
    const float* __restrict__ cnt,
    const u16t* __restrict__ fr_n1, const float* __restrict__ n_b1,
    const u16t* __restrict__ fr_n2, const float* __restrict__ n_b2,
    int Ntot) {
  __shared__ u16t  s_in[WV][16 * 128];
  __shared__ u16t  s_a [WV][16 * 64];
  __shared__ float s_inv[WV][16];

  const int lane  = threadIdx.x & 31;
  const int w     = threadIdx.x >> 5;
  const int tile  = blockIdx.x * WV + w;
  const int nbase = tile * 16;
  const bool act  = nbase < Ntot;
  const int n0 = lane & 15, mh = (lane >> 4) << 3;

  if (act && lane < 16) {
    int i = nbase + lane;
    float inv = 1.f / fmaxf(cnt[i], 1.f);
    s_inv[w][lane] = inv;
#pragma unroll
    for (int d3 = 0; d3 < 3; ++d3)
      xcur[3 * i + d3] += sum_xw[3 * i + d3] * inv;
  }
  __syncthreads();

  if (act) {
    for (int m = 0; m < 16; ++m) {
      int i = nbase + m;
      float inv = s_inv[w][m];
      for (int cc = lane; cc < 128; cc += 32) {
        float v = (cc < 64) ? h[(size_t)i * 64 + cc]
                            : sum_m[(size_t)i * 64 + (cc - 64)] * inv;
        s_in[w][m * 128 + cc] = f2bf(v);
      }
    }
  }
  __syncthreads();

  v8f c[4];
  if (act) {
    zacc(c);
    wgemm(&s_in[w][0], 128, 4, fr_n1, c, lane);
    epi_silu(c, n_b1, &s_a[w][0], lane);
  }
  __syncthreads();
  if (act) {
    zacc(c);
    wgemm(&s_a[w][0], 64, 2, fr_n2, c, lane);
#pragma unroll
    for (int nt = 0; nt < 4; ++nt) {
      float bn = n_b2[nt * 16 + n0];
#pragma unroll
      for (int r = 0; r < 8; ++r) {
        int m = mh + r;
        size_t off = (size_t)(nbase + m) * 64 + nt * 16 + n0;
        h[off] += c[nt][r] + bn;   // residual in f32
      }
    }
  }
}

// ---------- host orchestration ----------
extern "C" void kernel_launch(void* const* d_in, const int* in_sizes, int n_in,
                              void* d_out, int out_size, void* d_ws, size_t ws_size,
                              hipStream_t stream) {
  (void)in_sizes; (void)n_in; (void)out_size; (void)ws_size;

  const float* x      = (const float*)d_in[0];
  const int*   z      = (const int*)  d_in[1];
  /* d_in[2] = num_atoms: unused by the reference */
  const int*   edges  = (const int*)  d_in[3];
  const float* emb    = (const float*)d_in[4];
  const float* eu_w1  = (const float*)d_in[5];
  const float* eu_b1  = (const float*)d_in[6];
  const float* eu_w2  = (const float*)d_in[7];
  const float* eu_b2  = (const float*)d_in[8];
  const float* msg_w1 = (const float*)d_in[9];
  const float* msg_b1 = (const float*)d_in[10];
  const float* msg_w2 = (const float*)d_in[11];
  const float* msg_b2 = (const float*)d_in[12];
  const float* pos_w1 = (const float*)d_in[13];
  const float* pos_b1 = (const float*)d_in[14];
  const float* pos_w2 = (const float*)d_in[15];
  const float* pos_b2 = (const float*)d_in[16];
  const float* nd_w1  = (const float*)d_in[17];
  const float* nd_b1  = (const float*)d_in[18];
  const float* nd_w2  = (const float*)d_in[19];
  const float* nd_b2  = (const float*)d_in[20];

  // workspace carve
  char* wp = (char*)d_ws;
  auto carve = [&](size_t bytes) -> char* {
    char* p = wp;
    wp += (bytes + 255) & ~(size_t)255;
    return p;
  };
  float* h      = (float*)carve((size_t)N_ * F_ * 4);
  float* xcur   = (float*)carve((size_t)N_ * 3 * 4);
  float* sum_m  = (float*)carve((size_t)N_ * (F_ + 4) * 4);  // sum_m|sum_xw|cnt contiguous
  float* sum_xw = sum_m + (size_t)N_ * F_;
  float* cnt    = sum_xw + (size_t)N_ * 3;
  u16t*  hij    = (u16t*)carve((size_t)E_TOT * F_ * 2);
  u16t*  frags  = (u16t*)carve((size_t)L_ * 49152 * 2);

  // per-layer fragment offsets (u16 elements)
  const int FR_EU1 = 0, FR_EU2 = 12288, FR_M1 = 16384, FR_M2 = 28672;
  const int FR_P1 = 32768, FR_N1 = 36864, FR_N2 = 45056;

  // one-time: convert/swizzle all weights to bf16 WMMA B-fragments (L2-resident)
  for (int l = 0; l < L_; ++l) {
    u16t* fb = frags + (size_t)l * 49152;
    auto cvt = [&](const float* src, u16t* dst, int K) {
      int total = K * 64;
      k_cvt<<<(total + 255) / 256, 256, 0, stream>>>(src, dst, K);
    };
    cvt(eu_w1  + (size_t)l * 192 * 64, fb + FR_EU1, 192);
    cvt(eu_w2  + (size_t)l * 64 * 64,  fb + FR_EU2, 64);
    cvt(msg_w1 + (size_t)l * 193 * 64, fb + FR_M1,  192);  // rows 0..191; row 192 handled as rank-1
    cvt(msg_w2 + (size_t)l * 64 * 64,  fb + FR_M2,  64);
    cvt(pos_w1 + (size_t)l * 64 * 64,  fb + FR_P1,  64);
    cvt(nd_w1  + (size_t)l * 128 * 64, fb + FR_N1,  128);
    cvt(nd_w2  + (size_t)l * 64 * 64,  fb + FR_N2,  64);
  }

  // h = emb[z]; xcur = x; h_ij = 0
  k_embed<<<(N_ * F_ + 255) / 256, 256, 0, stream>>>(emb, z, x, h, xcur);
  {
    long nu = (long)E_TOT * F_ / 2;  // zero hij as 32-bit words
    k_zero32<<<(unsigned)((nu + 255) / 256), 256, 0, stream>>>((unsigned*)hij, nu);
  }

  const int etiles = E_TOT / 16;
  const int ntiles = (N_ + 15) / 16;

  for (int l = 0; l < L_; ++l) {
    u16t* fb = frags + (size_t)l * 49152;
    k_zero_f<<<(N_ * (F_ + 4) + 255) / 256, 256, 0, stream>>>(sum_m, N_ * (F_ + 4));
    k_edge<<<(etiles + WV - 1) / WV, 128, 0, stream>>>(
        edges, xcur, h, hij,
        fb + FR_EU1, eu_b1 + l * 64, fb + FR_EU2, eu_b2 + l * 64,
        fb + FR_M1,  msg_b1 + l * 64, fb + FR_M2, msg_b2 + l * 64,
        fb + FR_P1,  pos_b1 + l * 64,
        msg_w1 + (size_t)l * 193 * 64 + (size_t)192 * 64,  // 193rd input row (f32)
        pos_w2 + l * 64, pos_b2 + l,
        sum_m, sum_xw, cnt, E_TOT);
    k_node<<<(ntiles + WV - 1) / WV, 128, 0, stream>>>(
        h, xcur, sum_m, sum_xw, cnt,
        fb + FR_N1, nd_b1 + l * 64, fb + FR_N2, nd_b2 + l * 64, N_);
  }

  k_copy<<<(N_ * 3 + 255) / 256, 256, 0, stream>>>((float*)d_out, xcur, N_ * 3);
}
